// FConv2d_18622978195587
// MI455X (gfx1250) — matrix operand order repaired and checked by reading, not compiled, not run on hardware
//
#include <hip/hip_runtime.h>
#include <hip/hip_bf16.h>

// ---------------------------------------------------------------------------
// FConv2d via banded implicit-GEMM with CDNA5 WMMA (bf16 -> f32 accumulate)
//
//   out[b, cc*8+n, h, w] = sum_{c'<32, i,j<3} x[b, (4cc-c')%128, h+1-i, w+1-j]
//                                             * W[n, c', i, j]
// (spatial taps outside [0,64) are zeros -- they hit the FFT zero-pad line)
//
// GEMM mapping per WMMA tile:  A = weights (M=16 oc x K=32 ch),
//                              B = x patch (K=32 ch x N=16 w-cols),
// so D's lane index = w  =>  fully coalesced 64B output segments per store.
// Weight slab staged to LDS with GLOBAL_LOAD_ASYNC_TO_LDS_B128 (ASYNCcnt).
// ---------------------------------------------------------------------------

typedef __attribute__((ext_vector_type(16))) __bf16 v16bf;
typedef __attribute__((ext_vector_type(8)))  float  v8f;

#define LSZ   64
#define CINC  128
#define COUTC 256
#define TAPS  9                // 3x3
#define WEXP_K (TAPS * CINC)   // 1152 reduction length per output channel

// tiling
#define TILE_ROWS 8            // output rows per workgroup (one per wave)
#define TILE_COLS 16           // output cols per workgroup (= WMMA N)
#define HALO_ROWS (TILE_ROWS + 2)   // 10
#define HALO_COLS (TILE_COLS + 2)   // 18
#define OCG       64           // output channels per workgroup (4 WMMA M-tiles)

// LDS layout (bytes); dynamic-LDS only => region starts at LDS offset 0
#define WS_BYTES (OCG * WEXP_K * 2)                 // 147456: ws[ocl][t][c] bf16
#define XS_BYTES (HALO_ROWS * HALO_COLS * CINC * 2) // 46080 : xs[r][c][ch] bf16
#define SMEM_BYTES (WS_BYTES + XS_BYTES)            // 193536 < 320KB WGP LDS

// ---------------------------------------------------------------------------
// Prep: expand the 32-channel circular band into dense Wexp[oc][tap][cin] bf16
// ---------------------------------------------------------------------------
__global__ __launch_bounds__(256)
void wexp_build_kernel(const float* __restrict__ W, __bf16* __restrict__ Wexp) {
    int idx = blockIdx.x * 256 + threadIdx.x;
    if (idx >= COUTC * WEXP_K) return;
    int oc  = idx / WEXP_K;
    int rem = idx - oc * WEXP_K;
    int t   = rem / CINC;
    int c   = rem - t * CINC;
    int cc  = oc >> 3;
    int n   = oc & 7;
    int cp  = (4 * cc - c) & 127;            // circular channel offset, mod 128
    float v = 0.0f;
    if (cp < 32) {
        int i = t / 3, j = t - 3 * (t / 3);
        v = W[((n * 32 + cp) * 3 + i) * 3 + j];
    }
    Wexp[idx] = (__bf16)v;
}

// ---------------------------------------------------------------------------
// Main implicit-GEMM conv kernel
// grid: (32 position tiles, 4 oc groups, 8 batch), block: 256 (8 wave32)
// ---------------------------------------------------------------------------
__global__ __launch_bounds__(256)
void fconv_wmma_kernel(const float*  __restrict__ x,
                       const __bf16* __restrict__ Wexp,
                       float*        __restrict__ out) {
    extern __shared__ char smem[];
    __bf16* ws = (__bf16*)smem;               // [OCG][TAPS][CINC] @ LDS off 0
    __bf16* xs = (__bf16*)(smem + WS_BYTES);  // [HALO_ROWS][HALO_COLS][CINC]

    const int tid = threadIdx.x;
    const int b   = blockIdx.z;
    const int ocg = blockIdx.y;               // 0..3 -> oc base = ocg*64
    const int ty  = blockIdx.x >> 2;          // 0..7 -> rows ty*8..ty*8+7
    const int tx  = blockIdx.x & 3;           // 0..3 -> cols tx*16..tx*16+15

    // ---- stage weights: 144KB rectangular DMA via async-to-LDS b128 -------
    // LDS region is purely dynamic => ws begins at LDS byte offset 0, so the
    // global byte offset and LDS byte offset coincide.
    {
        const __bf16* wsrc = Wexp + (size_t)ocg * (OCG * WEXP_K);
        #pragma unroll 4
        for (int k = tid; k < WS_BYTES / 16; k += 256) {
            unsigned off = (unsigned)(k * 16);   // LDS offset == global voffset
            asm volatile(
                "global_load_async_to_lds_b128 %0, %1, %2"
                :: "v"(off), "v"(off), "s"(wsrc)
                : "memory");
        }
    }
    // ---- stage input halo, fp32 -> bf16 on the fly -------------------------
    {
        const int base_r = ty * TILE_ROWS - 1;
        const int base_c = tx * TILE_COLS - 1;
        for (int e = tid; e < HALO_ROWS * HALO_COLS * CINC; e += 256) {
            int ch = e / (HALO_ROWS * HALO_COLS);
            int s  = e - ch * (HALO_ROWS * HALO_COLS);
            int r  = s / HALO_COLS;
            int ci = s - r * HALO_COLS;
            int gr = base_r + r;
            int gc = base_c + ci;
            float v = 0.0f;
            if ((unsigned)gr < 64u && (unsigned)gc < 64u)
                v = x[(((size_t)b * CINC + ch) * LSZ + gr) * LSZ + gc];
            xs[s * CINC + ch] = (__bf16)v;
        }
    }
    asm volatile("s_wait_asynccnt 0" ::: "memory");
    __syncthreads();

    // ---- per-wave GEMM: M=4x16 oc, N=16 positions (one row), K=1152 --------
    const int wv   = tid >> 5;        // wave id == local output row (0..7)
    const int lane = tid & 31;
    const int ln   = lane & 15;       // A: M row (oc) ; B/D: N col (w)
    const int hif  = lane >> 4;       // half-wave select
    const int koffA = hif ? 8  : 0;   // 16-bit A layout: K 0-7/16-23 vs 8-15/24-31
    const int koffB = hif ? 8  : 0;   // 16-bit B layout: K 0-7/16-23 vs 8-15/24-31

    v8f acc[4] = {v8f{}, v8f{}, v8f{}, v8f{}};

    for (int t = 0; t < TAPS; ++t) {
        const int i = t / 3;
        const int j = t - 3 * i;
        const int hr = wv + 2 - i;            // halo row = (h+1-i) - (ty*8-1)
        const int hc = ln + 2 - j;            // halo col = (w+1-j) - (tx*16-1)
        const __bf16* xrow = xs + (hr * HALO_COLS + hc) * CINC;
        #pragma unroll
        for (int qb = 0; qb < 4; ++qb) {      // channel blocks of 32 (= WMMA K)
            // B fragment (x patch): channels koffB+0..7 and koffB+16..23 for column ln
            union { uint4 q[2]; v16bf v; } ub;
            {
                const uint4* pb = (const uint4*)(xrow + qb * 32 + koffB);
                ub.q[0] = pb[0];      // K 0-7   (or 8-15)
                ub.q[1] = pb[2];      // K 16-23 (or 24-31): +32 bytes
            }
            // 4 M-tiles of weights sharing this B fragment
            #pragma unroll
            for (int nt = 0; nt < 4; ++nt) {
                const int ocl = nt * 16 + ln; // A row = oc within group
                union { uint4 q[2]; v16bf v; } ua;
                const uint4* pa =
                    (const uint4*)(ws + (ocl * TAPS + t) * CINC + qb * 32 + koffA);
                ua.q[0] = pa[0];      // K 0-7   (or 8-15)
                ua.q[1] = pa[2];      // K 16-23 (or 24-31): +32 bytes
                acc[nt] = __builtin_amdgcn_wmma_f32_16x16x32_bf16(
                    false, ua.v, false, ub.v, (short)0, acc[nt], false, false);
            }
        }
    }

    // ---- epilogue: D[m=oc][n=w] -> out[b][oc][h][w], 64B-coalesced ---------
    const int h = ty * TILE_ROWS + wv;
    const int wcol = tx * TILE_COLS + ln;     // N index = lane&15 -> w
    #pragma unroll
    for (int nt = 0; nt < 4; ++nt) {
        #pragma unroll
        for (int v = 0; v < 8; ++v) {
            const int oc = ocg * OCG + nt * 16 + v + (hif ? 8 : 0);  // M index
            out[(((size_t)b * COUTC + oc) * LSZ + h) * LSZ + wcol] = acc[nt][v];
        }
    }
}

// ---------------------------------------------------------------------------
extern "C" void kernel_launch(void* const* d_in, const int* in_sizes, int n_in,
                              void* d_out, int out_size, void* d_ws, size_t ws_size,
                              hipStream_t stream) {
    const float* x = (const float*)d_in[0];   // (8, 128, 64, 64) f32
    const float* W = (const float*)d_in[1];   // (8, 32, 3, 3)    f32
    float* out = (float*)d_out;               // (8, 256, 64, 64) f32
    __bf16* wexp = (__bf16*)d_ws;             // 256*9*128 bf16 = 589,824 B

    int total = COUTC * WEXP_K;
    wexp_build_kernel<<<(total + 255) / 256, 256, 0, stream>>>(W, wexp);

    dim3 grid(32, COUTC / OCG, 8);            // pos tiles x oc groups x batch
    fconv_wmma_kernel<<<grid, 256, SMEM_BYTES, stream>>>(x, wexp, out);
}
